// SnapshotRetrieval_4501125726456
// MI455X (gfx1250) — matrix operand order repaired and checked by reading, not compiled, not run on hardware
//
#include <hip/hip_runtime.h>

// ---------------------------------------------------------------------------
// SnapshotRetrieval for MI455X (gfx1250): bf16 WMMA everywhere GEMM-shaped,
// LDS-staged B panels, async-to-LDS staging for the attention Q tile.
// ---------------------------------------------------------------------------

#define DEVINL __device__ __forceinline__

typedef __bf16 bf16;
typedef __bf16 v16bf __attribute__((ext_vector_type(16)));
typedef __bf16 v8bf  __attribute__((ext_vector_type(8)));
typedef float  v8f   __attribute__((ext_vector_type(8)));

static constexpr int Bc    = 2;
static constexpr int Tc    = 2048;
static constexpr int Dm    = 2048;
static constexpr int Nc    = 1024;
static constexpr int Hc    = 16;
static constexpr int Rc    = 512;
static constexpr int DHc   = 128;
static constexpr int MLATc = 128;
static constexpr int TOPKc = 64;

DEVINL bf16 f2bf(float f) {
  union { float f; unsigned u; } v; v.f = f;
  unsigned r = (v.u + 0x7FFFu + ((v.u >> 16) & 1u)) >> 16;
  unsigned short h = (unsigned short)r;
  bf16 o; __builtin_memcpy(&o, &h, 2);
  return o;
}
DEVINL float bf2f(bf16 x) {
  unsigned short h; __builtin_memcpy(&h, &x, 2);
  union { unsigned u; float f; } v; v.u = ((unsigned)h) << 16;
  return v.f;
}
DEVINL bf16 ld_bf(const float* p) { return f2bf(*p); }
DEVINL bf16 ld_bf(const bf16*  p) { return *p; }

DEVINL float wave_sum(float x) {
  for (int off = 16; off; off >>= 1) x += __shfl_xor(x, off, 32);
  return x;
}
DEVINL int wave_sum_i(int x) {
  for (int off = 16; off; off >>= 1) x += __shfl_xor(x, off, 32);
  return x;
}
DEVINL float wave_max(float x) {
  for (int off = 16; off; off >>= 1) x = fmaxf(x, __shfl_xor(x, off, 32));
  return x;
}

// ---------------------------------------------------------------------------
// Async global->LDS copy (16B per lane) + completion wait (gfx1250 path).
// Inline asm per the CDNA5 bridge doc (portable across both toolchains).
// LDS operand is the 32-bit LDS byte offset (low 32 bits of generic pointer).
// ---------------------------------------------------------------------------
DEVINL void async_ld16(void* lds, const void* g) {
  asm volatile("global_load_async_to_lds_b128 %0, %1, off"
               :
               : "v"((unsigned)(size_t)lds),
                 "v"((unsigned long long)(size_t)g)
               : "memory");
}
DEVINL void async_wait_all() {
#if __has_builtin(__builtin_amdgcn_s_wait_asynccnt)
  __builtin_amdgcn_s_wait_asynccnt(0);
#else
  asm volatile("s_wait_asynccnt 0" ::: "memory");
#endif
}

// ---------------------------------------------------------------------------
// Row-major GEMM: C[M,N] = A[M,K] * B[K,N], bf16 WMMA, f32 accum.
// Block = 128 threads (4 waves). Block tile 64(M) x 64(N); wave w owns rows
// [m0, m0+16) and all 64 columns -> 4 accumulators (A-fragment reused 4x).
// B panel (32 x 64 f32) is staged per k-step into LDS transposed + converted
// to bf16, so every WMMA B-fragment is two contiguous 16-byte ds_loads.
// Grid: (M/64, N/64).
// Lane layouts (ISA 7.12.2):
//   A 16x32 bf16 : lane l -> row l&15; half=l>>4 -> K {kb..kb+7, 16+kb..16+kb+7}
//   B 32x16 bf16 : lane l -> col l&15; half=l>>4 -> K {kb..kb+7, 16+kb..16+kb+7}
//   C 16x16 f32  : lane l -> col l&15; VGPR j -> row j + 8*half
// ---------------------------------------------------------------------------
template <typename AT>
__global__ __launch_bounds__(128)
void gemm_wmma_bf16(const AT* __restrict__ A, const float* __restrict__ Bm,
                    float* __restrict__ C, int M, int K, int Nn) {
  __shared__ __align__(32) bf16 BsT[64][32];   // [n][k], 4KB

  const int w    = threadIdx.x >> 5;
  const int l    = threadIdx.x & 31;
  const int m0   = blockIdx.x * 64 + w * 16;
  const int n0   = blockIdx.y * 64;
  const int row  = l & 15;
  const int half = l >> 4;

  v8f acc[4];
#pragma unroll
  for (int nt = 0; nt < 4; nt++)
    acc[nt] = (v8f){0.f, 0.f, 0.f, 0.f, 0.f, 0.f, 0.f, 0.f};

  const AT* arow = A + (size_t)(m0 + row) * K;

  for (int k0 = 0; k0 < K; k0 += 32) {
    __syncthreads();   // previous iteration's fragment reads are done
    // Stage B panel: 32x64 f32, coalesced in n; store transposed bf16.
#pragma unroll
    for (int e = threadIdx.x; e < 32 * 64; e += 128) {
      int kk = e >> 6, nn = e & 63;
      BsT[nn][kk] = f2bf(Bm[(size_t)(k0 + kk) * Nn + n0 + nn]);
    }
    __syncthreads();

    v16bf a;
    const AT* ap = arow + k0 + half * 8;
#pragma unroll
    for (int i = 0; i < 8; i++) {
      a[i]     = ld_bf(ap + i);
      a[8 + i] = ld_bf(ap + 16 + i);
    }
#pragma unroll
    for (int nt = 0; nt < 4; nt++) {
      v8bf blo = *(const v8bf*)&BsT[nt * 16 + row][half * 8];
      v8bf bhi = *(const v8bf*)&BsT[nt * 16 + row][16 + half * 8];
      v16bf b;
#pragma unroll
      for (int i = 0; i < 8; i++) { b[i] = blo[i]; b[8 + i] = bhi[i]; }
      acc[nt] = __builtin_amdgcn_wmma_f32_16x16x32_bf16(false, a, false, b,
                                                        (short)0, acc[nt],
                                                        false, false);
    }
  }

#pragma unroll
  for (int nt = 0; nt < 4; nt++)
#pragma unroll
    for (int j = 0; j < 8; j++)
      C[(size_t)(m0 + j + 8 * half) * Nn + n0 + nt * 16 + row] = acc[nt][j];
}

// ---------------------------------------------------------------------------
// RMS-norm + RoPE for Q. One block (128 thr) per flat row bid = ((b*T+t)*H+h).
// ---------------------------------------------------------------------------
__global__ __launch_bounds__(128)
void rope_q_kernel(const float* __restrict__ qpre, const float* __restrict__ qw,
                   const int* __restrict__ toko, bf16* __restrict__ qbf) {
  const int bid = blockIdx.x;
  const int h  = bid % Hc;
  const int bt = bid / Hc;
  const int t  = bt % Tc;
  const int b  = bt / Tc;
  const int i  = threadIdx.x;
  const int w  = i >> 5, l = i & 31;

  __shared__ float wsum[4];
  __shared__ float xs[DHc];

  float xv = qpre[(size_t)bid * DHc + i];
  float ss = wave_sum(xv * xv);
  if (l == 0) wsum[w] = ss;
  __syncthreads();
  float tot = wsum[0] + wsum[1] + wsum[2] + wsum[3];
  float rms = rsqrtf(tot / (float)DHc + 1e-6f);
  float xn  = xv * rms * qw[i];
  xs[i] = xn;
  __syncthreads();

  int   j   = i & 63;
  float inv = __powf(10000.0f, -(float)(2 * j) / (float)DHc);
  float ang = (float)(toko[0] + t) * inv;
  float rot = (i < 64) ? -xs[i + 64] : xs[i - 64];
  float o   = xn * __cosf(ang) + rot * __sinf(ang);
  qbf[((size_t)(b * Hc + h) * Tc + t) * DHc + i] = f2bf(o);
}

// ---------------------------------------------------------------------------
// RMS-norm + RoPE for K. bid = ((b*N+n)*H+h). pos = snap_positions[n].
// ---------------------------------------------------------------------------
__global__ __launch_bounds__(128)
void rope_k_kernel(const float* __restrict__ kpre, const float* __restrict__ kw,
                   const int* __restrict__ spos, bf16* __restrict__ kbf) {
  const int bid = blockIdx.x;
  const int h  = bid % Hc;
  const int bn = bid / Hc;
  const int n  = bn % Nc;
  const int b  = bn / Nc;
  const int i  = threadIdx.x;
  const int w  = i >> 5, l = i & 31;

  __shared__ float wsum[4];
  __shared__ float xs[DHc];

  float xv = kpre[(size_t)bid * DHc + i];
  float ss = wave_sum(xv * xv);
  if (l == 0) wsum[w] = ss;
  __syncthreads();
  float tot = wsum[0] + wsum[1] + wsum[2] + wsum[3];
  float rms = rsqrtf(tot / (float)DHc + 1e-6f);
  float xn  = xv * rms * kw[i];
  xs[i] = xn;
  __syncthreads();

  int   j   = i & 63;
  float inv = __powf(10000.0f, -(float)(2 * j) / (float)DHc);
  float ang = (float)spos[n] * inv;
  float rot = (i < 64) ? -xs[i + 64] : xs[i - 64];
  float o   = xn * __cosf(ang) + rot * __sinf(ang);
  kbf[((size_t)(b * Hc + h) * Nc + n) * DHc + i] = f2bf(o);
}

// ---------------------------------------------------------------------------
// V: (b,n,h,dh) f32 -> transposed bf16 (b,h,dh,n) so W*V B-fragments are
// two contiguous 16-byte loads each.
// ---------------------------------------------------------------------------
__global__ __launch_bounds__(256)
void conv_v_kernel(const float* __restrict__ vpre, bf16* __restrict__ vbfT) {
  size_t idx = (size_t)blockIdx.x * 256 + threadIdx.x;
  if (idx >= (size_t)Bc * Nc * Hc * DHc) return;
  int    dh = idx % DHc;
  size_t r  = idx / DHc;
  int    h  = r % Hc;
  size_t bn = r / Hc;
  int    n  = bn % Nc;
  int    b  = (int)(bn / Nc);
  vbfT[((size_t)(b * Hc + h) * DHc + dh) * Nc + n] = f2bf(vpre[idx]);
}

// ---------------------------------------------------------------------------
// Attention: one block (256 thr = 8 waves) per (b,h, 16-row t tile).
// Phase 0: Q tile (4KB) staged to LDS via global_load_async_to_lds_b128.
// Phase 1: scores S[16][1024] = (Q Kt)/sqrt(dh) via WMMA  (S stored bf16).
// Phase 2: per row: causal mask, top-64 threshold (binary search on
//          order-preserving uint keys, wave32 reductions), softmax with sink,
//          weights written in place (bf16).
// Phase 3: out[16][128] = W * V via WMMA; store f32 (b,t,h*dh).
// LDS: 32KB scores/weights + 4KB Q tile = 36KB.
// ---------------------------------------------------------------------------
__global__ __launch_bounds__(256)
void attn_kernel(const bf16* __restrict__ qbf, const bf16* __restrict__ kbf,
                 const bf16* __restrict__ vbfT, const int* __restrict__ spos,
                 const int* __restrict__ toko, const float* __restrict__ sinkl,
                 float* __restrict__ attn_out) {
  __shared__ __align__(32) bf16 S[16][Nc];    // scores, then weights (in place)
  __shared__ __align__(32) bf16 Qs[16][DHc];  // staged Q tile

  const int tiles_t = Tc / 16;
  const int bid = blockIdx.x;
  const int tt  = bid % tiles_t;
  const int bh  = bid / tiles_t;
  const int h   = bh % Hc;
  const int b   = bh / Hc;
  const int t0  = tt * 16;
  const int w   = threadIdx.x >> 5;
  const int l   = threadIdx.x & 31;
  const int col = l & 15;
  const int half = l >> 4;

  // Phase 0: async-stage Q tile (16 rows x 128 bf16 = 4KB; 16B per thread).
  {
    const bf16* qsrc = qbf + ((size_t)bh * Tc + t0) * DHc;
    const int e = threadIdx.x;              // 0..255, one b128 each
    async_ld16(&((bf16*)Qs)[e * 8], qsrc + e * 8);
    async_wait_all();
  }
  __syncthreads();

  // A fragments (Q rows) — same 16-row tile for every wave.
  v16bf afr[4];
#pragma unroll
  for (int c = 0; c < 4; c++) {
    int  k0 = c * 32;
    v8bf lo = *(const v8bf*)&Qs[col][k0 + half * 8];
    v8bf hi = *(const v8bf*)&Qs[col][k0 + 16 + half * 8];
#pragma unroll
    for (int i = 0; i < 8; i++) { afr[c][i] = lo[i]; afr[c][8 + i] = hi[i]; }
  }

  // Phase 1: scores. Each wave covers 8 of the 64 key tiles.
  const bf16* kbase = kbf + (size_t)bh * Nc * DHc;
  for (int nt = w; nt < Nc / 16; nt += 8) {
    int n0 = nt * 16;
    v8f acc = {0.f, 0.f, 0.f, 0.f, 0.f, 0.f, 0.f, 0.f};
#pragma unroll
    for (int c = 0; c < 4; c++) {
      int   k0 = c * 32;
      const bf16* kp = kbase + (size_t)(n0 + col) * DHc + k0 + half * 8;
      v8bf blo = *(const v8bf*)kp;
      v8bf bhi = *(const v8bf*)(kp + 16);
      v16bf bfrag;
#pragma unroll
      for (int i = 0; i < 8; i++) { bfrag[i] = blo[i]; bfrag[8 + i] = bhi[i]; }
      acc = __builtin_amdgcn_wmma_f32_16x16x32_bf16(false, afr[c], false, bfrag,
                                                    (short)0, acc, false, false);
    }
#pragma unroll
    for (int j = 0; j < 8; j++)
      S[j + 8 * half][n0 + col] = f2bf(acc[j] * 0.08838834764831845f);
  }
  __syncthreads();

  // Phase 2: mask + top-k + softmax. 2 rows per wave.
  const int tko = toko[0];
  for (int rr = 0; rr < 2; rr++) {
    const int rowi  = w * 2 + rr;
    const int t_abs = tko + t0 + rowi;

    float    v[32];
    unsigned key[32];
#pragma unroll
    for (int j = 0; j < 32; j++) {
      int   idx = l + 32 * j;
      float sv  = bf2f(S[rowi][idx]);
      if (spos[idx] >= t_abs) sv = -__builtin_inff();
      v[j] = sv;
      unsigned u; __builtin_memcpy(&u, &sv, 4);
      key[j] = (u & 0x80000000u) ? ~u : (u | 0x80000000u);
    }

    // Largest threshold u with count(key >= u) >= TOPK.
    unsigned thr = 0;
    for (int bit = 31; bit >= 0; --bit) {
      unsigned cand = thr | (1u << bit);
      int cnt = 0;
#pragma unroll
      for (int j = 0; j < 32; j++) cnt += (key[j] >= cand) ? 1 : 0;
      cnt = wave_sum_i(cnt);
      if (cnt >= TOPKc) thr = cand;
    }

    const float sink = sinkl[h];
    float mx = sink;
#pragma unroll
    for (int j = 0; j < 32; j++)
      if (key[j] >= thr && v[j] > mx) mx = v[j];
    mx = wave_max(mx);

    float e[32];
    float sum = 0.f;
#pragma unroll
    for (int j = 0; j < 32; j++) {
      float ev = (key[j] >= thr) ? __expf(v[j] - mx) : 0.f;
      e[j] = ev; sum += ev;
    }
    sum = wave_sum(sum);
    sum += __expf(sink - mx);
    float inv = 1.f / sum;
#pragma unroll
    for (int j = 0; j < 32; j++)
      S[rowi][l + 32 * j] = f2bf(e[j] * inv);   // weights, in place
  }
  __syncthreads();

  // Phase 3: out = W * V. Wave w -> dh tile [w*16, w*16+16).
  {
    const int   d0   = w * 16;
    const bf16* vrow = vbfT + ((size_t)bh * DHc + d0 + col) * Nc;
    v8f acc = {0.f, 0.f, 0.f, 0.f, 0.f, 0.f, 0.f, 0.f};
    for (int k0 = 0; k0 < Nc; k0 += 32) {
      v16bf a;
      v8bf lo = *(const v8bf*)&S[col][k0 + half * 8];
      v8bf hi = *(const v8bf*)&S[col][k0 + 16 + half * 8];
#pragma unroll
      for (int i = 0; i < 8; i++) { a[i] = lo[i]; a[8 + i] = hi[i]; }
      v8bf blo = *(const v8bf*)(vrow + k0 + half * 8);
      v8bf bhi = *(const v8bf*)(vrow + k0 + 16 + half * 8);
      v16bf bfrag;
#pragma unroll
      for (int i = 0; i < 8; i++) { bfrag[i] = blo[i]; bfrag[8 + i] = bhi[i]; }
      acc = __builtin_amdgcn_wmma_f32_16x16x32_bf16(false, a, false, bfrag,
                                                    (short)0, acc, false, false);
    }
#pragma unroll
    for (int j = 0; j < 8; j++)
      attn_out[((size_t)(b * Tc + t0 + j + 8 * half) * Hc + h) * DHc + d0 + col] = acc[j];
  }
}

// ---------------------------------------------------------------------------
// obf = bf16(attn * sigmoid(gpre)), flat over (B*T, H*DH).
// ---------------------------------------------------------------------------
__global__ __launch_bounds__(256)
void gate_mul_kernel(const float* __restrict__ attn, const float* __restrict__ g,
                     bf16* __restrict__ obf, int n) {
  int i = blockIdx.x * 256 + threadIdx.x;
  if (i < n) {
    float gv = 1.f / (1.f + __expf(-g[i]));
    obf[i] = f2bf(attn[i] * gv);
  }
}

// ---------------------------------------------------------------------------
extern "C" void kernel_launch(void* const* d_in, const int* in_sizes, int n_in,
                              void* d_out, int out_size, void* d_ws, size_t ws_size,
                              hipStream_t stream) {
  const float* x     = (const float*)d_in[0];
  const float* snaps = (const float*)d_in[1];
  const int*   spos  = (const int*)d_in[2];
  const int*   toko  = (const int*)d_in[3];
  const float* Wqd   = (const float*)d_in[4];
  const float* Wqu   = (const float*)d_in[5];
  const float* Wgd   = (const float*)d_in[6];
  const float* Wgu   = (const float*)d_in[7];
  const float* Wod   = (const float*)d_in[8];
  const float* Wou   = (const float*)d_in[9];
  const float* Wku   = (const float*)d_in[10];
  const float* Wvu   = (const float*)d_in[11];
  const float* qnw   = (const float*)d_in[12];
  const float* knw   = (const float*)d_in[13];
  const float* sinkl = (const float*)d_in[14];
  float* out = (float*)d_out;

  const int BT  = Bc * Tc;          // 4096
  const int BNH = Bc * Nc * Hc;     // 32768

  char* p = (char*)d_ws;
  auto alloc = [&](size_t bytes) {
    void* r = (void*)p;
    p += (bytes + 255) & ~(size_t)255;
    return r;
  };
  float* regA = (float*)alloc((size_t)BT * (Hc * DHc) * 4);   // q_pre -> attn_out
  float* regB = (float*)alloc((size_t)BT * (Hc * DHc) * 4);   // k_pre/v_pre -> g_pre
  bf16*  qbf  = (bf16*) alloc((size_t)Bc * Hc * Tc * DHc * 2);
  bf16*  kbf  = (bf16*) alloc((size_t)Bc * Hc * Nc * DHc * 2);
  bf16*  vbfT = (bf16*) alloc((size_t)Bc * Hc * Nc * DHc * 2);
  float* qlat = (float*)alloc((size_t)BT * MLATc * 4);
  float* glat = (float*)alloc((size_t)BT * MLATc * 4);
  float* olat = (float*)alloc((size_t)BT * MLATc * 4);
  bf16*  obf  = (bf16*) alloc((size_t)BT * (Hc * DHc) * 2);

  float* k_pre = regB;                                   // 32768 x 128
  float* v_pre = regB + (size_t)BNH * DHc;               // 32768 x 128

  auto gemm_f = [&](const float* A, const float* Bm, float* C, int M, int K, int N) {
    dim3 g(M / 64, N / 64);
    gemm_wmma_bf16<float><<<g, 128, 0, stream>>>(A, Bm, C, M, K, N);
  };
  auto gemm_b = [&](const bf16* A, const float* Bm, float* C, int M, int K, int N) {
    dim3 g(M / 64, N / 64);
    gemm_wmma_bf16<bf16><<<g, 128, 0, stream>>>(A, Bm, C, M, K, N);
  };

  // --- Q / gate latent projections
  gemm_f(x, Wqd, qlat, BT, Dm, MLATc);
  gemm_f(x, Wgd, glat, BT, Dm, MLATc);

  // --- Q up-projection, norm + RoPE
  gemm_f(qlat, Wqu, regA, BT, MLATc, Hc * DHc);
  rope_q_kernel<<<BT * Hc, 128, 0, stream>>>(regA, qnw, toko, qbf);

  // --- K / V up-projections, norm + RoPE / transpose-convert
  gemm_f(snaps, Wku, k_pre, BNH, Rc, DHc);
  gemm_f(snaps, Wvu, v_pre, BNH, Rc, DHc);
  rope_k_kernel<<<BNH, 128, 0, stream>>>(k_pre, knw, spos, kbf);
  {
    size_t tot = (size_t)Bc * Nc * Hc * DHc;
    conv_v_kernel<<<(unsigned)((tot + 255) / 256), 256, 0, stream>>>(v_pre, vbfT);
  }

  // --- gate up-projection (regB free now)
  gemm_f(glat, Wgu, regB, BT, MLATc, Hc * DHc);

  // --- attention (writes into regA; q_pre is dead)
  attn_kernel<<<Bc * Hc * (Tc / 16), 256, 0, stream>>>(qbf, kbf, vbfT, spos, toko,
                                                       sinkl, regA);

  // --- gate multiply, convert to bf16
  {
    int tot = BT * Hc * DHc;
    gate_mul_kernel<<<(tot + 255) / 256, 256, 0, stream>>>(regA, regB, obf, tot);
  }

  // --- output projections
  gemm_b(obf, Wod, olat, BT, Hc * DHc, MLATc);
  gemm_f(olat, Wou, out, BT, MLATc, Dm);
}
